// LabelNetworkModule_43997644980465
// MI455X (gfx1250) — matrix operand order repaired and checked by reading, not compiled, not run on hardware
//
#include <hip/hip_runtime.h>
#include <hip/hip_bf16.h>

#define DEVINL __device__ __forceinline__

typedef __attribute__((ext_vector_type(16))) _Float16 v16h;
typedef __attribute__((ext_vector_type(8)))  _Float16 v8h;
typedef __attribute__((ext_vector_type(8)))  float    v8f;
typedef __attribute__((ext_vector_type(4)))  unsigned int v4u;
typedef __attribute__((ext_vector_type(4)))  int v4i;
typedef __attribute__((ext_vector_type(8)))  int v8i;

// Problem dims (fixed by the reference)
constexpr int NB = 4096;  // batch
constexpr int NL = 4096;  // labels
constexpr int NF = 128;   // feature dim
constexpr int NH = 128;   // hidden
constexpr int NO = 64;    // out dim of layer 2

// GEMM tiling: block tile 256x128, 8 waves, each wave owns a 64x64 tile
// (4x4 WMMA accumulators). 16 WMMAs per 16 ds_load_b128 per K-step.
constexpr int BM = 256;
constexpr int BN = 128;
constexpr int BK = 32;   // == WMMA K for f16
constexpr int LDSK = 40; // padded LDS row stride in halves: 80B, 16B-aligned;
                         // 20-dword row offset -> 16 consecutive rows hit disjoint banks

#if defined(__has_builtin)
#  if __has_builtin(__builtin_amdgcn_tensor_load_to_lds)
#    define HAVE_TDM 1
#  else
#    define HAVE_TDM 0
#  endif
#else
#  define HAVE_TDM 0
#endif

DEVINL void wait_tensor0() {
#if defined(__has_builtin) && __has_builtin(__builtin_amdgcn_s_wait_tensorcnt)
  __builtin_amdgcn_s_wait_tensorcnt(0);
#else
  asm volatile("s_wait_tensorcnt 0x0" ::: "memory");
#endif
}

#if HAVE_TDM
// Issue one TDM 2D tile load: rows x 32 halves, line stride ldElems halves,
// into LDS at byte offset lds_byte with +16B padding per 64B line (stride 80B).
DEVINL void tdm_load_tile_f16(unsigned int lds_byte, const _Float16* gsrc,
                              unsigned int ldElems, unsigned int rows) {
  unsigned long long ga = (unsigned long long)(size_t)gsrc;
  v4u g0;
  g0.x = 1u;                                              // count=1 (valid), no gather
  g0.y = lds_byte;                                        // lds_addr
  g0.z = (unsigned int)ga;                                // global_addr[31:0]
  g0.w = ((unsigned int)(ga >> 32) & 0x01FFFFFFu) | 0x80000000u; // addr[56:32] | type=2
  v8i g1;
  // data_size=1 (2B) | pad_enable | pad_interval=3 (16 dwords) | pad_amount=3 (4 dwords)
  g1[0] = (int)((1u << 16) | (1u << 20) | (3u << 22) | (3u << 25));
  g1[1] = (int)((ldElems & 0xFFFFu) << 16);                              // tensor_dim0 lo16
  g1[2] = (int)(((ldElems >> 16) & 0xFFFFu) | ((rows & 0xFFFFu) << 16)); // dim0 hi | dim1 lo
  g1[3] = (int)(((rows >> 16) & 0xFFFFu) | (32u << 16));                 // dim1 hi | tile_dim0=32
  g1[4] = (int)(rows & 0xFFFFu);                                         // tile_dim1=rows, tile_dim2=0
  g1[5] = (int)ldElems;                                                  // tensor_dim0_stride lo32
  g1[6] = 0;
  g1[7] = 0;
  v4i z4 = {0, 0, 0, 0};
#if __clang_major__ >= 23
  v8i z8 = {0, 0, 0, 0, 0, 0, 0, 0};
  __builtin_amdgcn_tensor_load_to_lds(g0, g1, z4, z4, z8, 0);
#else
  __builtin_amdgcn_tensor_load_to_lds(g0, g1, z4, z4, 0);
#endif
}
#else
// Fallback: cooperative load (rows x 32 halves tile, 16B chunks).
DEVINL void coop_load_tile_f16(_Float16* lds, const _Float16* gsrc, int ldElems, int rows) {
  int nchunks = rows * 4;  // 4 x 16B per 64B row
  for (int c = threadIdx.x; c < nchunks; c += blockDim.x) {
    int row = c >> 2, q = c & 3;
    *(v8h*)(lds + row * LDSK + q * 8) =
        *(const v8h*)(gsrc + (size_t)row * ldElems + q * 8);
  }
}
#endif

// A-fragment (16x32 f16, M x K): lane m=lane&15; lanes16-31 take K-halves 8..15/24..31
DEVINL v16h ldA(const _Float16* s, int m0, int lane) {
  int m = m0 + (lane & 15);
  int kh = (lane & 16) ? 8 : 0;
  const _Float16* p = s + m * LDSK + kh;
  v8h lo = *(const v8h*)p;         // K = kh .. kh+7
  v8h hi = *(const v8h*)(p + 16);  // K = kh+16 .. kh+23
  return __builtin_shufflevector(lo, hi, 0, 1, 2, 3, 4, 5, 6, 7, 8, 9, 10, 11, 12, 13, 14, 15);
}
// B-fragment (32x16 f16, K x N), NT form: per-lane column n == row n of Q (contiguous K)
DEVINL v16h ldB(const _Float16* s, int n0, int lane) {
  int n = n0 + (lane & 15);
  int kb = (lane & 16) ? 8 : 0;
  const _Float16* p = s + n * LDSK + kb;
  v8h lo = *(const v8h*)p;        // K = kb .. kb+7
  v8h hi = *(const v8h*)(p + 16); // K = kb+16 .. kb+23
  return __builtin_shufflevector(lo, hi, 0, 1, 2, 3, 4, 5, 6, 7, 8, 9, 10, 11, 12, 13, 14, 15);
}

// NT-GEMM: C[M,N] = P[MxK] * Q[NxK]^T, f16 in, f32 accumulate.
// EPI 0: store f32. EPI 1: store f16 (acc>0 ? 1 : 0) (adjacency). EPI 3: f32 acc/(e1[m]+e1[n]).
template <int EPI>
__global__ __launch_bounds__(256, 1) void gemm_nt_wmma(const _Float16* __restrict__ P,
                                                       const _Float16* __restrict__ Q,
                                                       void* __restrict__ OutV, int M, int N,
                                                       int K, const float* __restrict__ e1) {
  __shared__ _Float16 As[2][BM * LDSK];  // 2 x 256 x 40 halves = 40 KB
  __shared__ _Float16 Bs[2][BN * LDSK];  // 2 x 128 x 40 halves = 20 KB
  const int bm = blockIdx.y * BM, bn = blockIdx.x * BN;
  const int wid = threadIdx.x >> 5, lane = threadIdx.x & 31;
  const int wm = (wid >> 1) * 64;  // 4 wave-rows x 64
  const int wn = (wid & 1) * 64;   // 2 wave-cols x 64

  v8f acc[4][4];
#pragma unroll
  for (int i = 0; i < 4; ++i)
#pragma unroll
    for (int j = 0; j < 4; ++j) acc[i][j] = (v8f){0, 0, 0, 0, 0, 0, 0, 0};

  // Prologue: fill buffer 0
#if HAVE_TDM
  if (wid == 0) {
    tdm_load_tile_f16((unsigned int)(size_t)&As[0][0], P + (size_t)bm * K, (unsigned)K, BM);
    tdm_load_tile_f16((unsigned int)(size_t)&Bs[0][0], Q + (size_t)bn * K, (unsigned)K, BN);
    wait_tensor0();
  }
#else
  coop_load_tile_f16(&As[0][0], P + (size_t)bm * K, K, BM);
  coop_load_tile_f16(&Bs[0][0], Q + (size_t)bn * K, K, BN);
#endif
  __syncthreads();

  int buf = 0;
  for (int kk = 0; kk < K; kk += BK) {
    const bool more = (kk + BK) < K;
#if HAVE_TDM
    if (more && wid == 0) {
      tdm_load_tile_f16((unsigned int)(size_t)&As[buf ^ 1][0],
                        P + (size_t)bm * K + (kk + BK), (unsigned)K, BM);
      tdm_load_tile_f16((unsigned int)(size_t)&Bs[buf ^ 1][0],
                        Q + (size_t)bn * K + (kk + BK), (unsigned)K, BN);
    }
#else
    if (more) {
      coop_load_tile_f16(&As[buf ^ 1][0], P + (size_t)bm * K + (kk + BK), K, BM);
      coop_load_tile_f16(&Bs[buf ^ 1][0], Q + (size_t)bn * K + (kk + BK), K, BN);
    }
#endif
    const _Float16* Ab = &As[buf][0];
    const _Float16* Bb = &Bs[buf][0];
    v16h a0 = ldA(Ab, wm, lane);
    v16h a1 = ldA(Ab, wm + 16, lane);
    v16h a2 = ldA(Ab, wm + 32, lane);
    v16h a3 = ldA(Ab, wm + 48, lane);
#pragma unroll
    for (int ni = 0; ni < 4; ++ni) {
      v16h b = ldB(Bb, wn + ni * 16, lane);
      acc[0][ni] = __builtin_amdgcn_wmma_f32_16x16x32_f16(false, a0, false, b, (short)0,
                                                          acc[0][ni], false, false);
      acc[1][ni] = __builtin_amdgcn_wmma_f32_16x16x32_f16(false, a1, false, b, (short)0,
                                                          acc[1][ni], false, false);
      acc[2][ni] = __builtin_amdgcn_wmma_f32_16x16x32_f16(false, a2, false, b, (short)0,
                                                          acc[2][ni], false, false);
      acc[3][ni] = __builtin_amdgcn_wmma_f32_16x16x32_f16(false, a3, false, b, (short)0,
                                                          acc[3][ni], false, false);
    }
#if HAVE_TDM
    if (more && wid == 0) wait_tensor0();
#endif
    __syncthreads();
    buf ^= 1;
  }

  // Epilogue. C/D layout: VGPR r holds M=r (lanes 0-15) / M=8+r (lanes 16-31), N=lane&15.
  const int mh = (lane & 16) ? 8 : 0;
  const int cn = lane & 15;
#pragma unroll
  for (int mi = 0; mi < 4; ++mi) {
#pragma unroll
    for (int ni = 0; ni < 4; ++ni) {
      const int gm0 = bm + wm + mi * 16 + mh;
      const int gn = bn + wn + ni * 16 + cn;
#pragma unroll
      for (int r = 0; r < 8; ++r) {
        const int gm = gm0 + r;
        const float v = acc[mi][ni][r];
        const size_t idx = (size_t)gm * (size_t)N + (size_t)gn;
        if (EPI == 1) {
          ((_Float16*)OutV)[idx] = (v > 0.0f) ? (_Float16)1.0f : (_Float16)0.0f;
        } else if (EPI == 3) {
          float d = e1[gm] + e1[gn];
          if (__builtin_fabsf(d) < 1e-8f) d = 1.0f;
          ((float*)OutV)[idx] = v / d;
        } else {
          ((float*)OutV)[idx] = v;
        }
      }
    }
  }
}

// ----- support kernels (negligible FLOPs) -----

__global__ void cast_f32_f16(const float* __restrict__ s, _Float16* __restrict__ d, size_t n) {
  size_t i = (size_t)blockIdx.x * blockDim.x + threadIdx.x;
  size_t stride = (size_t)gridDim.x * blockDim.x;
  for (; i < n; i += stride) d[i] = (_Float16)s[i];
}

// dst[C x R] (f16) = transpose(src[R x C] f32); R, C multiples of 32; block (32,8)
__global__ void transpose_f32_f16(const float* __restrict__ src, _Float16* __restrict__ dst,
                                  int R, int C) {
  __shared__ float t[32][33];
  int c0 = blockIdx.x * 32, r0 = blockIdx.y * 32;
#pragma unroll
  for (int j = 0; j < 32; j += 8)
    t[threadIdx.y + j][threadIdx.x] = src[(size_t)(r0 + threadIdx.y + j) * C + c0 + threadIdx.x];
  __syncthreads();
#pragma unroll
  for (int j = 0; j < 32; j += 8)
    dst[(size_t)(c0 + threadIdx.y + j) * R + r0 + threadIdx.x] =
        (_Float16)t[threadIdx.x][threadIdx.y + j];
}

// inv_deg[j] = 1/max(sum_k A[j][k], 1)  (A symmetric -> row sum == column sum)
__global__ void degree_kernel(const _Float16* __restrict__ A, float* __restrict__ invdeg, int L) {
  __shared__ float red[256];
  int j = blockIdx.x;
  float p = 0.f;
  for (int i = threadIdx.x; i < L; i += 256) p += (float)A[(size_t)j * L + i];
  red[threadIdx.x] = p;
  __syncthreads();
  for (int s = 128; s > 0; s >>= 1) {
    if (threadIdx.x < (unsigned)s) red[threadIdx.x] += red[threadIdx.x + s];
    __syncthreads();
  }
  if (threadIdx.x == 0) invdeg[j] = 1.0f / fmaxf(red[0], 1.0f);
}

// out[l,n] = act( invdeg[l]*sum_k agg[l,k]*Wl[k,n] + bl[n] + sum_k x[l,k]*Wr[k,n] )
__global__ void sage_dense(const float* __restrict__ agg, const float* __restrict__ x,
                           const float* __restrict__ invdeg, const float* __restrict__ Wl,
                           const float* __restrict__ bl, const float* __restrict__ Wr,
                           float* __restrict__ out, int KIN, int NOUT, int relu) {
  __shared__ float sa[128], sx[128];
  int l = blockIdx.x;
  for (int k = threadIdx.x; k < KIN; k += blockDim.x) {
    sa[k] = agg[(size_t)l * KIN + k];
    sx[k] = x[(size_t)l * KIN + k];
  }
  __syncthreads();
  float id = invdeg[l];
  for (int n = threadIdx.x; n < NOUT; n += blockDim.x) {
    float s1 = 0.f, s2 = 0.f;
    for (int k = 0; k < KIN; ++k) {
      s1 += sa[k] * Wl[k * NOUT + n];
      s2 += sx[k] * Wr[k * NOUT + n];
    }
    float v = s1 * id + s2 + bl[n];
    out[(size_t)l * NOUT + n] = relu ? fmaxf(v, 0.f) : v;
  }
}

__global__ void gate_raw(const float* __restrict__ y2, const float* __restrict__ Wg,
                         const float* __restrict__ bg, float* __restrict__ raw, int L, int O) {
  int l = blockIdx.x * blockDim.x + threadIdx.x;
  if (l < L) {
    float a = bg[0];
    for (int k = 0; k < O; ++k) a += y2[(size_t)l * O + k] * Wg[k];
    raw[l] = tanhf(a);
  }
}

__global__ void minmax_kernel(const float* __restrict__ raw, int n, float* __restrict__ mm) {
  __shared__ float smn[1024], smx[1024];
  float mn = 1e30f, mx = -1e30f;
  for (int i = threadIdx.x; i < n; i += blockDim.x) {
    float v = raw[i];
    mn = fminf(mn, v);
    mx = fmaxf(mx, v);
  }
  smn[threadIdx.x] = mn;
  smx[threadIdx.x] = mx;
  __syncthreads();
  for (int s = 512; s > 0; s >>= 1) {
    if (threadIdx.x < (unsigned)s) {
      smn[threadIdx.x] = fminf(smn[threadIdx.x], smn[threadIdx.x + s]);
      smx[threadIdx.x] = fmaxf(smx[threadIdx.x], smx[threadIdx.x + s]);
    }
    __syncthreads();
  }
  if (threadIdx.x == 0) {
    mm[0] = smn[0];
    mm[1] = smx[0];
  }
}

// lw[b,l] = f16(l_t[b,l]*omega[l]); s[b] = sum_l lw row (f32)
__global__ void weight_rows(const float* __restrict__ l_t, const float* __restrict__ raw,
                            const float* __restrict__ mm, _Float16* __restrict__ lw,
                            float* __restrict__ s, int L) {
  __shared__ float red[256];
  int b = blockIdx.x;
  float mn = mm[0], mx = mm[1];
  bool flat = __builtin_fabsf(mx - mn) < 1e-8f;
  float rng = flat ? 1.0f : (mx - mn);
  float partial = 0.f;
  for (int l = threadIdx.x; l < L; l += 256) {
    float om = flat ? 0.5f : (raw[l] - mn) / rng;
    float v = l_t[(size_t)b * L + l] * om;
    lw[(size_t)b * L + l] = (_Float16)v;
    partial += v;
  }
  red[threadIdx.x] = partial;
  __syncthreads();
  for (int st = 128; st > 0; st >>= 1) {
    if (threadIdx.x < (unsigned)st) red[threadIdx.x] += red[threadIdx.x + st];
    __syncthreads();
  }
  if (threadIdx.x == 0) s[b] = red[0];
}

extern "C" void kernel_launch(void* const* d_in, const int* in_sizes, int n_in, void* d_out,
                              int out_size, void* d_ws, size_t ws_size, hipStream_t stream) {
  const float* l_t = (const float*)d_in[0];
  const float* y0  = (const float*)d_in[1];
  const float* W1l = (const float*)d_in[2];
  const float* b1l = (const float*)d_in[3];
  const float* W1r = (const float*)d_in[4];
  const float* W2l = (const float*)d_in[5];
  const float* b2l = (const float*)d_in[6];
  const float* W2r = (const float*)d_in[7];
  const float* Wg  = (const float*)d_in[8];
  const float* bg  = (const float*)d_in[9];
  float* out = (float*)d_out;

  char* ws = (char*)d_ws;
  size_t off = 0;
  auto take = [&](size_t bytes) -> char* {
    char* p = ws + off;
    off = (off + bytes + 255) & ~(size_t)255;
    return p;
  };
  _Float16* X16   = (_Float16*)take((size_t)NB * NL * 2);  // l_t in f16
  _Float16* Xt16  = (_Float16*)take((size_t)NL * NB * 2);  // l_t^T in f16
  _Float16* Aadj  = (_Float16*)take((size_t)NL * NL * 2);  // adjacency 0/1 f16
  _Float16* lw16  = (_Float16*)take((size_t)NB * NL * 2);  // weighted rows f16
  float*    invdg = (float*)take((size_t)NL * 4);
  float*    aggf  = (float*)take((size_t)NL * NF * 4);     // reused for layer 2
  float*    y1    = (float*)take((size_t)NL * NH * 4);
  _Float16* xtbuf = (_Float16*)take((size_t)NH * NL * 2);  // x^T f16 (layer1 then layer2)
  float*    y2    = (float*)take((size_t)NL * NO * 4);
  float*    raw   = (float*)take((size_t)NL * 4);
  float*    mm    = (float*)take(256);
  float*    sbuf  = (float*)take((size_t)NB * 4);

  // 1) f16 copies of l_t and l_t^T
  cast_f32_f16<<<4096, 256, 0, stream>>>(l_t, X16, (size_t)NB * NL);
  transpose_f32_f16<<<dim3(NL / 32, NB / 32), dim3(32, 8), 0, stream>>>(l_t, Xt16, NB, NL);

  // 2) Gram + threshold: Aadj = (Xt * Xt^T > 0)
  gemm_nt_wmma<1><<<dim3(NL / BN, NL / BM), 256, 0, stream>>>(Xt16, Xt16, (void*)Aadj, NL, NL,
                                                              NB, nullptr);
  // 3) degrees
  degree_kernel<<<NL, 256, 0, stream>>>(Aadj, invdg, NL);

  // 4) layer 1: agg = A @ y0  (NT form with y0^T)
  transpose_f32_f16<<<dim3(NF / 32, NL / 32), dim3(32, 8), 0, stream>>>(y0, xtbuf, NL, NF);
  gemm_nt_wmma<0><<<dim3(NF / BN, NL / BM), 256, 0, stream>>>(Aadj, xtbuf, (void*)aggf, NL, NF,
                                                              NL, nullptr);
  sage_dense<<<NL, 128, 0, stream>>>(aggf, y0, invdg, W1l, b1l, W1r, y1, NF, NH, 1);

  // 5) layer 2
  transpose_f32_f16<<<dim3(NH / 32, NL / 32), dim3(32, 8), 0, stream>>>(y1, xtbuf, NL, NH);
  gemm_nt_wmma<0><<<dim3(NH / BN, NL / BM), 256, 0, stream>>>(Aadj, xtbuf, (void*)aggf, NL, NH,
                                                              NL, nullptr);
  sage_dense<<<NL, 128, 0, stream>>>(aggf, y1, invdg, W2l, b2l, W2r, y2, NH, NO, 0);

  // 6) gate -> omega, weighted rows, row sums
  gate_raw<<<NL / 256, 256, 0, stream>>>(y2, Wg, bg, raw, NL, NO);
  minmax_kernel<<<1, 1024, 0, stream>>>(raw, NL, mm);
  weight_rows<<<NB, 256, 0, stream>>>(l_t, raw, mm, lw16, sbuf, NL);

  // 7) numerator / (s_i + s_j) fused
  gemm_nt_wmma<3><<<dim3(NB / BN, NB / BM), 256, 0, stream>>>(lw16, X16, (void*)out, NB, NB, NL,
                                                              sbuf);
  (void)in_sizes; (void)n_in; (void)out_size; (void)ws_size;
}
